// MambaDynamicsModel_37082747633898
// MI455X (gfx1250) — matrix operand order repaired and checked by reading, not compiled, not run on hardware
//
#include <hip/hip_runtime.h>
#include <hip/hip_bf16.h>

// ---------------------------------------------------------------------------
// Mamba dynamics model on gfx1250 (MI455X).
// GEMMs: bf16 WMMA (v_wmma_f32_16x16x32_bf16), f32 accumulation.
// Selective scan: 3-pass chunked associative scan (16 chunks x 256 steps).
// ---------------------------------------------------------------------------

namespace {
constexpr int kB   = 4;
constexpr int kL   = 4096;
constexpr int kSD  = 3;
constexpr int kDIM = 128;
constexpr int kNL  = 4;
constexpr int kDS  = 16;
constexpr int kDC  = 4;
constexpr int kDI  = 256;   // 2*DIM
constexpr int kDTR = 8;
constexpr int kNT  = kB * kL;         // 16384 tokens
constexpr int kNCH = 16;              // scan chunks per sequence
constexpr int kLC  = kL / kNCH;       // 256 steps per chunk

// workspace byte offsets
constexpr size_t H_OFF     = 0;                                          // h      (NT,128) f32
constexpr size_t XZ_OFF    = H_OFF    + (size_t)kNT * kDIM * 4;          // xz     (NT,512) f32
constexpr size_t XC_OFF    = XZ_OFF   + (size_t)kNT * 2 * kDI * 4;       // xc     (NT,256) f32
constexpr size_t DEL_OFF   = XC_OFF   + (size_t)kNT * kDI * 4;           // delta  (NT,256) f32
constexpr size_t BM_OFF    = DEL_OFF  + (size_t)kNT * kDI * 4;           // Bm     (NT,16)  f32
constexpr size_t CM_OFF    = BM_OFF   + (size_t)kNT * kDS * 4;           // Cm     (NT,16)  f32
constexpr size_t YZ_OFF    = CM_OFF   + (size_t)kNT * kDS * 4;           // yz     (NT,256) f32
constexpr size_t AP_OFF    = YZ_OFF   + (size_t)kNT * kDI * 4;           // aprod  (B,NCH,DI,DS)
constexpr size_t HL_OFF    = AP_OFF   + (size_t)kB*kNCH*kDI*kDS * 4;     // hloc
constexpr size_t HI_OFF    = HL_OFF   + (size_t)kB*kNCH*kDI*kDS * 4;     // hinit
constexpr size_t WTIN_OFF  = HI_OFF   + (size_t)kB*kNCH*kDI*kDS * 4;     // in_proj bf16 [l][n=512][k=128]
constexpr size_t WTOUT_OFF = WTIN_OFF + (size_t)kNL * (2*kDI) * kDIM * 2;// out_proj bf16 [l][n=128][k=256]
} // namespace

typedef __attribute__((ext_vector_type(8)))  float  v8f;
typedef __attribute__((ext_vector_type(16))) __bf16 v16bf;
typedef __attribute__((ext_vector_type(8)))  __bf16 v8bf;

__device__ __forceinline__ float silu_f(float x)     { return x / (1.0f + __expf(-x)); }
__device__ __forceinline__ float softplus_f(float x) { return (x > 20.0f) ? x : log1pf(__expf(x)); }
__device__ __forceinline__ float gelu_f(float x)     { return 0.5f * x * (1.0f + erff(x * 0.70710678118654752f)); }

// ---------------------------------------------------------------------------
// Weight conversion: fp32 -> bf16, transposed to [N][K] so each WMMA lane's
// B-fragment is one contiguous 32-byte load.
// ---------------------------------------------------------------------------
__global__ void mamba_convert_weights(const float* __restrict__ in_w,
                                      const float* __restrict__ out_w,
                                      __bf16* __restrict__ wtin,
                                      __bf16* __restrict__ wtout) {
  int idx = blockIdx.x * blockDim.x + threadIdx.x;
  if (idx < kNL * 2 * kDI * kDIM) {            // 4*512*128
    int l = idx / (2 * kDI * kDIM);
    int r = idx % (2 * kDI * kDIM);
    int n = r / kDIM;
    int k = r % kDIM;
    wtin[idx] = (__bf16)in_w[((size_t)l * kDIM + k) * (2 * kDI) + n];
  }
  if (idx < kNL * kDIM * kDI) {                // 4*128*256
    int l = idx / (kDIM * kDI);
    int r = idx % (kDIM * kDI);
    int n = r / kDI;
    int k = r % kDI;
    wtout[idx] = (__bf16)out_w[((size_t)l * kDI + k) * kDIM + n];
  }
}

// ---------------------------------------------------------------------------
// Embedding: h = gelu(x @ W1 + b1) @ W2 + b2      (K=3 then 128x128, VALU)
// ---------------------------------------------------------------------------
__global__ void mamba_embed(const float* __restrict__ x,
                            const float* __restrict__ w1, const float* __restrict__ b1,
                            const float* __restrict__ w2, const float* __restrict__ b2,
                            float* __restrict__ h) {
  __shared__ float t1[kDIM];
  int tok = blockIdx.x;
  int d   = threadIdx.x;
  float s = b1[d];
#pragma unroll
  for (int k = 0; k < kSD; ++k) s += x[tok * kSD + k] * w1[k * kDIM + d];
  t1[d] = gelu_f(s);
  __syncthreads();
  float o = b2[d];
#pragma unroll 8
  for (int j = 0; j < kDIM; ++j) o += t1[j] * w2[j * kDIM + d];
  h[(size_t)tok * kDIM + d] = o;
}

// ---------------------------------------------------------------------------
// RMSNorm + in_proj GEMM (128 -> 512), bf16 WMMA. 16 tokens / block, 8 waves,
// each wave owns 4 N-tiles of 16. K loop: 4 x 32.
// ---------------------------------------------------------------------------
__global__ void mamba_rms_inproj(const float* __restrict__ h,
                                 const float* __restrict__ rms_w,
                                 const __bf16* __restrict__ wt,   // [512][128] bf16
                                 float* __restrict__ xz) {
  __shared__ float hs[16 * kDIM];
  __shared__ float rinv[16];
  __shared__ alignas(32) __bf16 hb[16 * kDIM];

  const int tid  = threadIdx.x;
  const int row0 = blockIdx.x * 16;

  for (int idx = tid; idx < 16 * kDIM; idx += 256)
    hs[idx] = h[(size_t)row0 * kDIM + idx];
  __syncthreads();
  if (tid < 16) {
    float s = 0.f;
#pragma unroll 8
    for (int k = 0; k < kDIM; ++k) { float v = hs[tid * kDIM + k]; s += v * v; }
    rinv[tid] = rsqrtf(s * (1.0f / kDIM) + 1e-5f);
  }
  __syncthreads();
  for (int idx = tid; idx < 16 * kDIM; idx += 256) {
    int t = idx >> 7, k = idx & 127;
    hb[idx] = (__bf16)(hs[idx] * rinv[t] * rms_w[k]);
  }
  __syncthreads();

  const int wave = tid >> 5;
  const int lane = tid & 31;
  const int m    = lane & 15;
  const int khiA = (lane >= 16) ? 8 : 0;
  const int khiB = (lane >= 16) ? 16 : 0;
  const int ncol = lane & 15;

  v8f acc[4];
#pragma unroll
  for (int t = 0; t < 4; ++t)
#pragma unroll
    for (int r = 0; r < 8; ++r) acc[t][r] = 0.0f;

#pragma unroll
  for (int ks = 0; ks < 4; ++ks) {
    const int kb = ks * 32;
    v8bf lo = *(const v8bf*)&hb[m * kDIM + kb + khiB];
    v8bf hi = *(const v8bf*)&hb[m * kDIM + kb + khiB + 8];
    v16bf afrag;
#pragma unroll
    for (int i = 0; i < 8; ++i) { afrag[i] = lo[i]; afrag[i + 8] = hi[i]; }
#pragma unroll
    for (int t = 0; t < 4; ++t) {
      int n = (wave * 4 + t) * 16 + ncol;
      v16bf bfrag = *(const v16bf*)&wt[n * kDIM + kb + khiB];
      acc[t] = __builtin_amdgcn_wmma_f32_16x16x32_bf16(
          false, afrag, false, bfrag, (short)0, acc[t], false, false);
    }
  }

  const int mbase = (lane >= 16) ? 8 : 0;
#pragma unroll
  for (int t = 0; t < 4; ++t) {
    int n = (wave * 4 + t) * 16 + ncol;
#pragma unroll
    for (int r = 0; r < 8; ++r)
      xz[(size_t)(row0 + mbase + r) * (2 * kDI) + n] = acc[t][r];
  }
}

// ---------------------------------------------------------------------------
// Causal depthwise conv (4 taps) + SiLU + x_proj (256->40) + dt_proj (8->256)
// + softplus; emits xc, delta, Bm, Cm. One token per block, 256 threads.
// ---------------------------------------------------------------------------
__global__ void mamba_conv_xproj(const float* __restrict__ xz,
                                 const float* __restrict__ conv_w,
                                 const float* __restrict__ conv_b,
                                 const float* __restrict__ xproj_w,  // [256][40]
                                 const float* __restrict__ dtproj_w, // [8][256]
                                 const float* __restrict__ dtproj_b,
                                 float* __restrict__ xc,
                                 float* __restrict__ delta,
                                 float* __restrict__ Bm,
                                 float* __restrict__ Cm) {
  __shared__ float xcs[kDI];
  __shared__ float dbc[kDTR + 2 * kDS];

  const int tok = blockIdx.x;
  const int b   = tok / kL;
  const int ll  = tok % kL;
  const int e   = threadIdx.x;

  float a = conv_b[e];
#pragma unroll
  for (int k = 0; k < kDC; ++k) {
    int lp = ll - (kDC - 1) + k;
    if (lp >= 0) a += xz[((size_t)(b * kL + lp)) * (2 * kDI) + e] * conv_w[e * kDC + k];
  }
  a = silu_f(a);
  xcs[e] = a;
  xc[(size_t)tok * kDI + e] = a;
  __syncthreads();

  if (e < kDTR + 2 * kDS) {
    float s = 0.f;
#pragma unroll 8
    for (int j = 0; j < kDI; ++j) s += xcs[j] * xproj_w[j * (kDTR + 2 * kDS) + e];
    dbc[e] = s;
  }
  __syncthreads();

  float s = dtproj_b[e];
#pragma unroll
  for (int r = 0; r < kDTR; ++r) s += dbc[r] * dtproj_w[r * kDI + e];
  delta[(size_t)tok * kDI + e] = softplus_f(s);

  if (e < kDS)                 Bm[(size_t)tok * kDS + e]        = dbc[kDTR + e];
  else if (e < 2 * kDS)        Cm[(size_t)tok * kDS + (e - kDS)] = dbc[kDTR + kDS + (e - kDS)];
}

// ---------------------------------------------------------------------------
// Scan pass A: per-chunk carry (cumulative dA product + local end state).
// Block: 256 thr = 16 channels x 16 states. Grid: B * NCH * (DI/16) = 1024.
// ---------------------------------------------------------------------------
__global__ void mamba_scan_carry(const float* __restrict__ delta,
                                 const float* __restrict__ xc,
                                 const float* __restrict__ Bm,
                                 const float* __restrict__ A_log,
                                 float* __restrict__ aprod,
                                 float* __restrict__ hloc) {
  const int gid = blockIdx.x;
  const int b   = gid / (kNCH * 16);
  const int rem = gid % (kNCH * 16);
  const int c   = rem / 16;
  const int eg  = rem % 16;
  const int n   = threadIdx.x & 15;
  const int e   = eg * 16 + (threadIdx.x >> 4);

  const float A = -__expf(A_log[e * kDS + n]);
  float ap = 1.0f, hl = 0.0f;
  const int lbase = c * kLC;
  for (int s = 0; s < kLC; ++s) {
    const int t = b * kL + lbase + s;
    __builtin_prefetch(&delta[(size_t)(t + 8) * kDI + e], 0, 1);
    const float dlt = delta[(size_t)t * kDI + e];
    const float xv  = xc[(size_t)t * kDI + e];
    const float bn  = Bm[(size_t)t * kDS + n];
    const float dA  = __expf(dlt * A);
    ap *= dA;
    hl = hl * dA + dlt * xv * bn;
  }
  const size_t o = ((size_t)(b * kNCH + c) * kDI + e) * kDS + n;
  aprod[o] = ap;
  hloc[o]  = hl;
}

// ---------------------------------------------------------------------------
// Scan pass B: stitch chunk carries sequentially (16 steps per lane).
// ---------------------------------------------------------------------------
__global__ void mamba_scan_stitch(const float* __restrict__ aprod,
                                  const float* __restrict__ hloc,
                                  float* __restrict__ hinit) {
  const int idx = blockIdx.x * blockDim.x + threadIdx.x;   // B*DI*DS = 16384
  const int b   = idx / (kDI * kDS);
  const int rem = idx % (kDI * kDS);
  const int e   = rem / kDS;
  const int n   = rem % kDS;
  float run = 0.0f;
#pragma unroll
  for (int c = 0; c < kNCH; ++c) {
    const size_t o = ((size_t)(b * kNCH + c) * kDI + e) * kDS + n;
    hinit[o] = run;
    run = run * aprod[o] + hloc[o];
  }
}

// ---------------------------------------------------------------------------
// Scan pass C: replay chunk with correct init state; fuse y = sum_n C*h + xc*D
// and z gating -> yz. width-16 shfl_xor reduction over states (wave32).
// ---------------------------------------------------------------------------
__global__ void mamba_scan_apply(const float* __restrict__ delta,
                                 const float* __restrict__ xc,
                                 const float* __restrict__ Bm,
                                 const float* __restrict__ Cm,
                                 const float* __restrict__ xz,     // z = xz[...,256:512]
                                 const float* __restrict__ A_log,
                                 const float* __restrict__ Dv,
                                 const float* __restrict__ hinit,
                                 float* __restrict__ yz) {
  const int gid = blockIdx.x;
  const int b   = gid / (kNCH * 16);
  const int rem = gid % (kNCH * 16);
  const int c   = rem / 16;
  const int eg  = rem % 16;
  const int n   = threadIdx.x & 15;
  const int e   = eg * 16 + (threadIdx.x >> 4);

  const float A  = -__expf(A_log[e * kDS + n]);
  const float Dp = Dv[e];
  float hst = hinit[((size_t)(b * kNCH + c) * kDI + e) * kDS + n];
  const int lbase = c * kLC;
  for (int s = 0; s < kLC; ++s) {
    const int t = b * kL + lbase + s;
    __builtin_prefetch(&delta[(size_t)(t + 8) * kDI + e], 0, 1);
    const float dlt = delta[(size_t)t * kDI + e];
    const float xv  = xc[(size_t)t * kDI + e];
    const float bn  = Bm[(size_t)t * kDS + n];
    const float cn  = Cm[(size_t)t * kDS + n];
    const float dA  = __expf(dlt * A);
    hst = hst * dA + dlt * xv * bn;
    float yv = hst * cn;
    yv += __shfl_xor(yv, 8, 16);
    yv += __shfl_xor(yv, 4, 16);
    yv += __shfl_xor(yv, 2, 16);
    yv += __shfl_xor(yv, 1, 16);
    if (n == 0) {
      const float zv = xz[(size_t)t * (2 * kDI) + kDI + e];
      yz[(size_t)t * kDI + e] = (yv + xv * Dp) * silu_f(zv);
    }
  }
}

// ---------------------------------------------------------------------------
// out_proj GEMM (256 -> 128) + residual add, bf16 WMMA. 16 tokens / block,
// 8 waves x one N-tile of 16. K loop: 8 x 32.
// ---------------------------------------------------------------------------
__global__ void mamba_outproj(const float* __restrict__ yz,
                              const __bf16* __restrict__ wt,   // [128][256] bf16
                              float* __restrict__ h) {
  __shared__ alignas(32) __bf16 ab[16 * kDI];

  const int tid  = threadIdx.x;
  const int row0 = blockIdx.x * 16;
  for (int idx = tid; idx < 16 * kDI; idx += 256)
    ab[idx] = (__bf16)yz[(size_t)row0 * kDI + idx];
  __syncthreads();

  const int wave = tid >> 5;
  const int lane = tid & 31;
  const int m    = lane & 15;
  const int khiA = (lane >= 16) ? 8 : 0;
  const int khiB = (lane >= 16) ? 16 : 0;
  const int n    = wave * 16 + (lane & 15);

  v8f acc;
#pragma unroll
  for (int r = 0; r < 8; ++r) acc[r] = 0.0f;

#pragma unroll
  for (int ks = 0; ks < 8; ++ks) {
    const int kb = ks * 32;
    v8bf lo = *(const v8bf*)&ab[m * kDI + kb + khiB];
    v8bf hi = *(const v8bf*)&ab[m * kDI + kb + khiB + 8];
    v16bf afrag;
#pragma unroll
    for (int i = 0; i < 8; ++i) { afrag[i] = lo[i]; afrag[i + 8] = hi[i]; }
    v16bf bfrag = *(const v16bf*)&wt[n * kDI + kb + khiB];
    acc = __builtin_amdgcn_wmma_f32_16x16x32_bf16(
        false, afrag, false, bfrag, (short)0, acc, false, false);
  }

  const int mbase = (lane >= 16) ? 8 : 0;
#pragma unroll
  for (int r = 0; r < 8; ++r) {
    const size_t o = (size_t)(row0 + mbase + r) * kDIM + n;
    h[o] += acc[r];
  }
}

// ---------------------------------------------------------------------------
// Final LayerNorm + head (128 -> 3).
// ---------------------------------------------------------------------------
__global__ void mamba_final(const float* __restrict__ h,
                            const float* __restrict__ nw, const float* __restrict__ nb,
                            const float* __restrict__ hw, const float* __restrict__ hb,
                            float* __restrict__ out) {
  __shared__ float sh[kDIM];
  __shared__ float red[kDIM];
  const int tok = blockIdx.x;
  const int d   = threadIdx.x;
  const float v = h[(size_t)tok * kDIM + d];
  red[d] = v;
  __syncthreads();
  for (int off = 64; off > 0; off >>= 1) {
    if (d < off) red[d] += red[d + off];
    __syncthreads();
  }
  const float mu = red[0] * (1.0f / kDIM);
  __syncthreads();
  const float cdev = v - mu;
  red[d] = cdev * cdev;
  __syncthreads();
  for (int off = 64; off > 0; off >>= 1) {
    if (d < off) red[d] += red[d + off];
    __syncthreads();
  }
  const float var = red[0] * (1.0f / kDIM);
  const float rinv = rsqrtf(var + 1e-5f);
  sh[d] = cdev * rinv * nw[d] + nb[d];
  __syncthreads();
  if (d < kSD) {
    float s = hb[d];
#pragma unroll 8
    for (int j = 0; j < kDIM; ++j) s += sh[j] * hw[j * kSD + d];
    out[(size_t)tok * kSD + d] = s;
  }
}

// ---------------------------------------------------------------------------
extern "C" void kernel_launch(void* const* d_in, const int* in_sizes, int n_in,
                              void* d_out, int out_size, void* d_ws, size_t ws_size,
                              hipStream_t stream) {
  (void)in_sizes; (void)n_in; (void)out_size; (void)ws_size;

  const float* x        = (const float*)d_in[0];
  const float* ew1      = (const float*)d_in[1];
  const float* eb1      = (const float*)d_in[2];
  const float* ew2      = (const float*)d_in[3];
  const float* eb2      = (const float*)d_in[4];
  const float* rms_w    = (const float*)d_in[5];
  const float* inproj_w = (const float*)d_in[6];
  const float* conv_w   = (const float*)d_in[7];
  const float* conv_b   = (const float*)d_in[8];
  const float* xproj_w  = (const float*)d_in[9];
  const float* dtproj_w = (const float*)d_in[10];
  const float* dtproj_b = (const float*)d_in[11];
  const float* A_log    = (const float*)d_in[12];
  const float* Dv       = (const float*)d_in[13];
  const float* outproj_w= (const float*)d_in[14];
  const float* norm_w   = (const float*)d_in[15];
  const float* norm_b   = (const float*)d_in[16];
  const float* head_w   = (const float*)d_in[17];
  const float* head_b   = (const float*)d_in[18];
  float* out = (float*)d_out;

  char* ws = (char*)d_ws;
  float*  h     = (float*)(ws + H_OFF);
  float*  xz    = (float*)(ws + XZ_OFF);
  float*  xc    = (float*)(ws + XC_OFF);
  float*  delta = (float*)(ws + DEL_OFF);
  float*  Bmv   = (float*)(ws + BM_OFF);
  float*  Cmv   = (float*)(ws + CM_OFF);
  float*  yz    = (float*)(ws + YZ_OFF);
  float*  aprod = (float*)(ws + AP_OFF);
  float*  hloc  = (float*)(ws + HL_OFF);
  float*  hinit = (float*)(ws + HI_OFF);
  __bf16* wtin  = (__bf16*)(ws + WTIN_OFF);
  __bf16* wtout = (__bf16*)(ws + WTOUT_OFF);

  mamba_convert_weights<<<(kNL * 2 * kDI * kDIM + 255) / 256, 256, 0, stream>>>(
      inproj_w, outproj_w, wtin, wtout);

  mamba_embed<<<kNT, kDIM, 0, stream>>>(x, ew1, eb1, ew2, eb2, h);

  for (int l = 0; l < kNL; ++l) {
    mamba_rms_inproj<<<kNT / 16, 256, 0, stream>>>(
        h, rms_w + l * kDIM, wtin + (size_t)l * 2 * kDI * kDIM, xz);

    mamba_conv_xproj<<<kNT, kDI, 0, stream>>>(
        xz, conv_w + l * kDI * kDC, conv_b + l * kDI,
        xproj_w + l * kDI * (kDTR + 2 * kDS),
        dtproj_w + l * kDTR * kDI, dtproj_b + l * kDI,
        xc, delta, Bmv, Cmv);

    mamba_scan_carry<<<kB * kNCH * (kDI / 16), 256, 0, stream>>>(
        delta, xc, Bmv, A_log + l * kDI * kDS, aprod, hloc);

    mamba_scan_stitch<<<(kB * kDI * kDS) / 256, 256, 0, stream>>>(aprod, hloc, hinit);

    mamba_scan_apply<<<kB * kNCH * (kDI / 16), 256, 0, stream>>>(
        delta, xc, Bmv, Cmv, xz, A_log + l * kDI * kDS, Dv + l * kDI, hinit, yz);

    mamba_outproj<<<kNT / 16, 256, 0, stream>>>(
        yz, wtout + (size_t)l * kDIM * kDI, h);
  }

  mamba_final<<<kNT, kDIM, 0, stream>>>(h, norm_w, norm_b, head_w, head_b, out);
}
